// LFQQuantizer_69483980915242
// MI455X (gfx1250) — matrix operand order repaired and checked by reading, not compiled, not run on hardware
//
#include <hip/hip_runtime.h>

// ---------------------------------------------------------------------------
// VQ nearest-codebook lookup for MI455X (gfx1250, wave32, WMMA).
//   cross = z_e @ codebook^T   via V_WMMA_F32_16X16X32_BF16 with bf16 hi/lo
//   split (x ~= hi + lo): cross ~= hi*hi + hi*lo + lo*hi  (fp32-class accuracy,
//   ~2.7x the matrix-pipe rate of the fp32 16x16x4 WMMA path).
//   d2 = |z|^2 - 2*cross + |c|^2, fused running argmin over K, then gather.
//   B tiles staged to LDS with GLOBAL_LOAD_ASYNC_TO_LDS_B128 (ASYNCcnt) into
//   a double buffer so the copy overlaps the WMMA burst.
// ---------------------------------------------------------------------------

typedef __attribute__((ext_vector_type(8)))  float  v8f;
typedef __attribute__((ext_vector_type(4)))  __bf16 v4bf;
typedef __attribute__((ext_vector_type(8)))  __bf16 v8bf;
typedef __attribute__((ext_vector_type(16))) __bf16 v16bf;
typedef __attribute__((ext_vector_type(4)))  int    v4i;

// async-to-LDS builtin wants pointer-to-int4 with explicit address spaces
typedef v4i __attribute__((address_space(1))) as1_v4i;
typedef v4i __attribute__((address_space(3))) as3_v4i;

#define B_N 8192
#define K_N 8192
#define D_N 512
#define SEGS 8
#define SEG_K (K_N / SEGS)   // 1024 codebook entries per segment
#define WG_M 64              // rows per workgroup (4 waves x 16)
#define NG 64                // codebook entries per LDS n-group (4 N-tiles)
#define DSTAGE 64            // D-depth staged per LDS round (2 WMMA K-steps)
#define BSTRIDE 72           // padded LDS row stride in bf16 elements (16B aligned)
#define NSTAGE ((SEG_K / NG) * (D_N / DSTAGE))   // 128 stages per block

__device__ __forceinline__ v16bf cat8(v8bf a, v8bf b) {
  return __builtin_shufflevector(a, b, 0,1,2,3,4,5,6,7,8,9,10,11,12,13,14,15);
}

// ---------------- prep: fp32 -> bf16 hi/lo split + squared row norms --------
__global__ __launch_bounds__(128) void vq_prep_kernel(
    const float* __restrict__ z_e, const float* __restrict__ codebook,
    __bf16* __restrict__ a_hi, __bf16* __restrict__ a_lo,
    __bf16* __restrict__ c_hi, __bf16* __restrict__ c_lo,
    float* __restrict__ z_sq, float* __restrict__ c_sq) {
  const int t = threadIdx.x;
  const int blk = blockIdx.x;
  const float* src;
  __bf16 *dhi, *dlo;
  float* sq;
  if (blk < K_N) {
    size_t r = (size_t)blk;
    src = codebook + r * D_N; dhi = c_hi + r * D_N; dlo = c_lo + r * D_N;
    sq = c_sq + blk;
  } else {
    size_t r = (size_t)(blk - K_N);
    src = z_e + r * D_N; dhi = a_hi + r * D_N; dlo = a_lo + r * D_N;
    sq = z_sq + (blk - K_N);
  }
  float4 x = ((const float4*)src)[t];          // 128 threads x 4 = 512 elems
  float xv[4] = {x.x, x.y, x.z, x.w};
  __bf16 hv[4], lv[4];
  float s = 0.0f;
#pragma unroll
  for (int i = 0; i < 4; ++i) {
    float xi = xv[i];
    __bf16 hb = (__bf16)xi;
    float  hf = (float)hb;
    hv[i] = hb;
    lv[i] = (__bf16)(xi - hf);
    s += xi * xi;
  }
  v4bf hvec = {hv[0], hv[1], hv[2], hv[3]};
  v4bf lvec = {lv[0], lv[1], lv[2], lv[3]};
  ((v4bf*)dhi)[t] = hvec;
  ((v4bf*)dlo)[t] = lvec;
  for (int off = 16; off > 0; off >>= 1) s += __shfl_xor(s, off, 32);
  __shared__ float red[4];
  if ((t & 31) == 0) red[t >> 5] = s;
  __syncthreads();
  if (t == 0) *sq = red[0] + red[1] + red[2] + red[3];
}

// ---------------- main: WMMA GEMM + fused per-segment argmin ----------------
__global__ __launch_bounds__(128) void vq_argmin_kernel(
    const __bf16* __restrict__ a_hi, const __bf16* __restrict__ a_lo,
    const __bf16* __restrict__ c_hi, const __bf16* __restrict__ c_lo,
    const float* __restrict__ z_sq, const float* __restrict__ c_sq,
    float* __restrict__ part_d, int* __restrict__ part_i) {
  // double-buffered B tiles (hi/lo): 2 x 2 x 64 x 72 bf16 = 36 KB LDS
  __shared__ __bf16 sHi[2][NG * BSTRIDE];
  __shared__ __bf16 sLo[2][NG * BSTRIDE];

  const int tid  = threadIdx.x;
  const int wave = tid >> 5;
  const int lane = tid & 31;
  const int lx   = lane & 15;     // C/D: n = lx
  const int lh   = lane >> 4;     // A/B lane-half selector
  const int m0   = blockIdx.x * WG_M + wave * 16;
  const int n00  = blockIdx.y * SEG_K;

  // issue async B-tile copy for stage sidx into buffer buf (8 b128/thread)
  auto stage_fn = [&](int sidx, int buf) {
    const int n0    = n00 + (sidx >> 3) * NG;
    const int dbase = (sidx & 7) * DSTAGE;
    for (int c = tid; c < (NG * DSTAGE) / 8; c += 128) {   // 4 iters/thread
      const int e   = c >> 3;                              // 8 chunks per row
      const int off = (c & 7) * 8;
      const size_t g = (size_t)(n0 + e) * D_N + dbase + off;
      __builtin_amdgcn_global_load_async_to_lds_b128(
          (as1_v4i*)&c_hi[g], (as3_v4i*)&sHi[buf][e * BSTRIDE + off], 0, 0);
      __builtin_amdgcn_global_load_async_to_lds_b128(
          (as1_v4i*)&c_lo[g], (as3_v4i*)&sLo[buf][e * BSTRIDE + off], 0, 0);
    }
  };

  float zs[8];
#pragma unroll
  for (int v = 0; v < 8; ++v) zs[v] = z_sq[m0 + v + 8 * lh];

  float best_d[8];
  int   best_i[8];
#pragma unroll
  for (int v = 0; v < 8; ++v) { best_d[v] = 3.4e38f; best_i[v] = 0; }

  v8f acc[4] = {};

  stage_fn(0, 0);                                   // prime the pipeline
  asm volatile("s_wait_asynccnt 0x0" ::: "memory");
  __syncthreads();

  for (int sidx = 0; sidx < NSTAGE; ++sidx) {
    const int buf   = sidx & 1;
    const int st    = sidx & 7;
    const int n0    = n00 + (sidx >> 3) * NG;
    const int dbase = st * DSTAGE;

    // kick off next stage's async copy into the other buffer; it drains
    // in the background of the 24-WMMA burst below.
    if (sidx + 1 < NSTAGE) {
      stage_fn(sidx + 1, buf ^ 1);
      const int dnext = ((sidx + 1) & 7) * DSTAGE;
      const size_t ap = (size_t)(m0 + lx) * D_N + dnext + 8 * lh;
      __builtin_prefetch(&a_hi[ap], 0, 3);          // global_prefetch_b8
      __builtin_prefetch(&a_lo[ap], 0, 3);
    }

#pragma unroll
    for (int dsl = 0; dsl < 2; ++dsl) {             // two 32-deep WMMA K-steps
      // A fragment (ISA 7.12.2): lane holds row m=lx, K = 8h..8h+7 and
      // 16+8h..16+8h+7 relative to the 32-wide chunk.
      const size_t abase =
          (size_t)(m0 + lx) * D_N + dbase + dsl * 32 + 8 * lh;
      v16bf Ah = cat8(*(const v8bf*)&a_hi[abase],
                      *(const v8bf*)&a_hi[abase + 16]);
      v16bf Al = cat8(*(const v8bf*)&a_lo[abase],
                      *(const v8bf*)&a_lo[abase + 16]);
#pragma unroll
      for (int j = 0; j < 4; ++j) {
        // B fragment: lane holds column n=j*16+lx, K = 8h..8h+7 and 16+8h..16+8h+7
        const int boff = (j * 16 + lx) * BSTRIDE + dsl * 32 + 8 * lh;
        v16bf Bh = cat8(*(const v8bf*)&sHi[buf][boff],
                        *(const v8bf*)&sHi[buf][boff + 16]);
        v16bf Bl = cat8(*(const v8bf*)&sLo[buf][boff],
                        *(const v8bf*)&sLo[buf][boff + 16]);
        acc[j] = __builtin_amdgcn_wmma_f32_16x16x32_bf16(
            false, Ah, false, Bh, (short)0, acc[j], false, false);
        acc[j] = __builtin_amdgcn_wmma_f32_16x16x32_bf16(
            false, Ah, false, Bl, (short)0, acc[j], false, false);
        acc[j] = __builtin_amdgcn_wmma_f32_16x16x32_bf16(
            false, Al, false, Bh, (short)0, acc[j], false, false);
      }
    }

    if (st == 7) {
      // epilogue: d2 = |z|^2 - 2*cross + |c|^2, running argmin; reset acc
#pragma unroll
      for (int j = 0; j < 4; ++j) {
        const int n = n0 + j * 16 + lx;
        const float cs = c_sq[n];
#pragma unroll
        for (int v = 0; v < 8; ++v) {
          float d2 = zs[v] - 2.0f * acc[j][v] + cs;
          if (d2 < best_d[v]) { best_d[v] = d2; best_i[v] = n; }
        }
      }
      v8f z = {};
#pragma unroll
      for (int j = 0; j < 4; ++j) acc[j] = z;
    }

    if (sidx + 1 < NSTAGE) {
      // drain this wave's outstanding async copies (next buffer), then make
      // them visible to all waves / release the buffer we just read.
      asm volatile("s_wait_asynccnt 0x0" ::: "memory");
      __syncthreads();
    }
  }

  // cross-lane argmin within each 16-lane half (lane L holds n = L%16 set)
#pragma unroll
  for (int v = 0; v < 8; ++v) {
    float bd = best_d[v];
    int   bi = best_i[v];
    for (int off = 8; off >= 1; off >>= 1) {
      float od = __shfl_xor(bd, off, 32);
      int   oi = __shfl_xor(bi, off, 32);
      if (od < bd || (od == bd && oi < bi)) { bd = od; bi = oi; }
    }
    if (lx == 0) {
      const int m = m0 + v + 8 * lh;
      part_d[(size_t)m * SEGS + blockIdx.y] = bd;
      part_i[(size_t)m * SEGS + blockIdx.y] = bi;
    }
  }
}

// ---------------- finalize: reduce 8 segment partials, gather z_q -----------
__global__ __launch_bounds__(128) void vq_finalize_kernel(
    const float* __restrict__ part_d, const int* __restrict__ part_i,
    const float* __restrict__ codebook,
    float* __restrict__ out_zq, float* __restrict__ out_idx) {
  const int row = blockIdx.x;
  __shared__ int sIdx;
  if (threadIdx.x == 0) {
    float bd = part_d[(size_t)row * SEGS];
    int   bi = part_i[(size_t)row * SEGS];
    for (int s = 1; s < SEGS; ++s) {
      float d = part_d[(size_t)row * SEGS + s];
      int   i = part_i[(size_t)row * SEGS + s];
      if (d < bd || (d == bd && i < bi)) { bd = d; bi = i; }
    }
    sIdx = bi;
    out_idx[row] = (float)bi;   // harness views output as float32
  }
  __syncthreads();
  const int bi = sIdx;
  const float4* src = (const float4*)(codebook + (size_t)bi * D_N);
  float4*       dst = (float4*)(out_zq + (size_t)row * D_N);
  dst[threadIdx.x] = src[threadIdx.x];          // 128 threads x 4 = 512
}

// ---------------------------------------------------------------------------
extern "C" void kernel_launch(void* const* d_in, const int* in_sizes, int n_in,
                              void* d_out, int out_size, void* d_ws,
                              size_t ws_size, hipStream_t stream) {
  const float* z_e = (const float*)d_in[0];   // [8192, 512] fp32
  const float* cb  = (const float*)d_in[1];   // [8192, 512] fp32

  // workspace layout (all offsets 256B aligned); total ~32.6 MiB
  char* ws = (char*)d_ws;
  const size_t MAT = (size_t)B_N * D_N * sizeof(__bf16);  // 8 MiB
  __bf16* a_hi = (__bf16*)(ws);
  __bf16* a_lo = (__bf16*)(ws + MAT);
  __bf16* c_hi = (__bf16*)(ws + 2 * MAT);
  __bf16* c_lo = (__bf16*)(ws + 3 * MAT);
  float*  z_sq = (float*)(ws + 4 * MAT);
  float*  c_sq = (float*)(ws + 4 * MAT + B_N * sizeof(float));
  float*  part_d = (float*)(ws + 4 * MAT + (B_N + K_N) * sizeof(float));
  int*    part_i = (int*)(ws + 4 * MAT + (B_N + K_N) * sizeof(float) +
                          (size_t)B_N * SEGS * sizeof(float));

  float* out_zq  = (float*)d_out;
  float* out_idx = (float*)d_out + (size_t)B_N * D_N;

  vq_prep_kernel<<<K_N + B_N, 128, 0, stream>>>(z_e, cb, a_hi, a_lo, c_hi,
                                                c_lo, z_sq, c_sq);
  dim3 grid(B_N / WG_M, SEGS);
  vq_argmin_kernel<<<grid, 128, 0, stream>>>(a_hi, a_lo, c_hi, c_lo, z_sq,
                                             c_sq, part_d, part_i);
  vq_finalize_kernel<<<B_N, 128, 0, stream>>>(part_d, part_i, cb, out_zq,
                                              out_idx);
}
